// InnerSoftShiftTripleModule_48009144435463
// MI455X (gfx1250) — matrix operand order repaired and checked by reading, not compiled, not run on hardware
//
#include <hip/hip_runtime.h>
#include <hip/hip_bf16.h>
#include <stdint.h>

// ---------------------------------------------------------------------------
// Problem constants (from reference): input (8, 512, 64, 64) f32
// ---------------------------------------------------------------------------
#define BZ    8
#define CTOT  512
#define CH    256
#define HW    4096          // 64*64
#define NM    1024          // hole positions (32x32)
#define NN    3072          // non-hole positions
#define OUTC  768
#define SPLIT 4             // split-K over n dimension for occupancy
#define NSEG  (NN / SPLIT)  // 768 columns per split
#define NTILES (BZ * (NM / 16))  // 512 (b, mtile) tiles

typedef __bf16 v16bf __attribute__((ext_vector_type(16)));
typedef __bf16 v8bf  __attribute__((ext_vector_type(8)));
typedef float  v8f   __attribute__((ext_vector_type(8)));

// hole is rows/cols [16,48)
__device__ __forceinline__ int mpos(int m) {
    return (16 + (m >> 5)) * 64 + 16 + (m & 31);
}
__device__ __forceinline__ int npos(int n) {
    if (n < 1024) return n;                       // rows 0..15 full
    if (n < 2048) {                               // rows 16..47, cols 0..15 & 48..63
        int i = n - 1024;
        int r = 16 + (i >> 5);
        int j = i & 31;
        int col = (j < 16) ? j : (j + 32);
        return r * 64 + col;
    }
    return n + 1024;                              // rows 48..63 full
}

__device__ __forceinline__ unsigned short f2bf(float x) {
    unsigned int u = __float_as_uint(x);
    u += 0x7FFFu + ((u >> 16) & 1u);              // round-to-nearest-even
    return (unsigned short)(u >> 16);
}

// ---------------------------------------------------------------------------
// Kernel 1: copy former+latter to out channels [0,512), zero shift [512,768)
// ---------------------------------------------------------------------------
__global__ void prep_copy_zero(const float* __restrict__ in, float* __restrict__ out) {
    const int64_t total = (int64_t)BZ * OUTC * (HW / 4);
    for (int64_t i = (int64_t)blockIdx.x * blockDim.x + threadIdx.x; i < total;
         i += (int64_t)gridDim.x * blockDim.x) {
        int p4 = (int)(i % (HW / 4));
        int64_t t = i / (HW / 4);
        int ch = (int)(t % OUTC);
        int b  = (int)(t / OUTC);
        float4 v = make_float4(0.f, 0.f, 0.f, 0.f);
        if (ch < CTOT)
            v = ((const float4*)in)[((int64_t)(b * CTOT + ch)) * (HW / 4) + p4];
        ((float4*)out)[i] = v;
    }
}

// ---------------------------------------------------------------------------
// Kernel 2a/2b: inverse channel norms (1 / max(||x||, 1e-8))
// ---------------------------------------------------------------------------
__global__ void fm_inv_norm(const float* __restrict__ in, float* __restrict__ invn) {
    int tid = blockIdx.x * blockDim.x + threadIdx.x;
    if (tid >= BZ * NM) return;
    int b = tid / NM, m = tid % NM;
    const float* p = in + (int64_t)b * CTOT * HW + mpos(m);
    float s = 0.f;
    for (int c = 0; c < CH; ++c) { float v = p[(int64_t)c * HW]; s += v * v; }
    invn[tid] = 1.0f / fmaxf(sqrtf(s), 1e-8f);
}

__global__ void ln_inv_norm(const float* __restrict__ in, float* __restrict__ invn) {
    int tid = blockIdx.x * blockDim.x + threadIdx.x;
    if (tid >= BZ * NN) return;
    int b = tid / NN, n = tid % NN;
    const float* p = in + ((int64_t)b * CTOT + CH) * HW + npos(n);
    float s = 0.f;
    for (int c = 0; c < CH; ++c) { float v = p[(int64_t)c * HW]; s += v * v; }
    invn[tid] = 1.0f / fmaxf(sqrtf(s), 1e-8f);
}

// ---------------------------------------------------------------------------
// Kernel 3: stage bf16 operands in WMMA-friendly layouts
//   Qa[b][m][c] = fm_n (normalized)      -> A operand of GEMM1
//   Kt[b][n][c] = ln_n (normalized)      -> B operand of GEMM1 (halves = c run)
//   Vt[b][c][n] = ln   (unnormalized)    -> B operand of GEMM2 (halves = n run)
// ---------------------------------------------------------------------------
__global__ void stage_q(const float* __restrict__ in, const float* __restrict__ invfm,
                        unsigned short* __restrict__ Qa) {
    const int64_t total = (int64_t)BZ * NM * CH;
    for (int64_t i = (int64_t)blockIdx.x * blockDim.x + threadIdx.x; i < total;
         i += (int64_t)gridDim.x * blockDim.x) {
        int c = (int)(i % CH);
        int m = (int)((i / CH) % NM);
        int b = (int)(i / ((int64_t)CH * NM));
        float v = in[((int64_t)(b * CTOT + c)) * HW + mpos(m)] * invfm[b * NM + m];
        Qa[i] = f2bf(v);
    }
}

__global__ void stage_k(const float* __restrict__ in, const float* __restrict__ invln,
                        unsigned short* __restrict__ Kt) {
    const int64_t total = (int64_t)BZ * NN * CH;
    for (int64_t i = (int64_t)blockIdx.x * blockDim.x + threadIdx.x; i < total;
         i += (int64_t)gridDim.x * blockDim.x) {
        int c = (int)(i % CH);
        int n = (int)((i / CH) % NN);
        int b = (int)(i / ((int64_t)CH * NN));
        float v = in[((int64_t)(b * CTOT + CH + c)) * HW + npos(n)] * invln[b * NN + n];
        Kt[i] = f2bf(v);
    }
}

__global__ void stage_v(const float* __restrict__ in, unsigned short* __restrict__ Vt) {
    const int64_t total = (int64_t)BZ * CH * NN;
    for (int64_t i = (int64_t)blockIdx.x * blockDim.x + threadIdx.x; i < total;
         i += (int64_t)gridDim.x * blockDim.x) {
        int n = (int)(i % NN);
        int c = (int)((i / NN) % CH);
        int b = (int)(i / ((int64_t)NN * CH));
        float v = in[((int64_t)(b * CTOT + CH + c)) * HW + npos(n)];
        Vt[i] = f2bf(v);
    }
}

// ---------------------------------------------------------------------------
// Kernel 4: fused flash attention, split-K over n (SPLIT partials / tile).
// One wave (32 threads) per (b, mtile, split). Emits partial accumulator and
// per-row (max, sum) statistics; kernel 5 merges them.
// ---------------------------------------------------------------------------
union BFU { v16bf v; v8bf h[2]; };

__global__ __launch_bounds__(32)
void flash_attn_part(const unsigned short* __restrict__ Qa,  // [8][1024][256]
                     const unsigned short* __restrict__ Kt,  // [8][3072][256]
                     const unsigned short* __restrict__ Vt,  // [8][256][3072]
                     float* __restrict__ accbuf,             // [SPLIT][512][16][256]
                     float* __restrict__ statbuf)            // [SPLIT][512][32]
{
    const int lane  = threadIdx.x;
    const int mtile = blockIdx.x;   // 0..63
    const int b     = blockIdx.y;   // 0..7
    const int split = blockIdx.z;   // 0..SPLIT-1
    const int tile  = b * 64 + mtile;
    const int half  = lane >> 4;    // 0|1
    const int l15   = lane & 15;
    const int koff  = half ? 16 : 0;

    __shared__ unsigned short pt16[16 * 32]; // P tile bf16 (transpose bounce)

    // --- Q fragments: A operand 16x32, contiguous k order as in K (lane=row; halves K {0..15}/{16..31})
    v16bf qf[8];
    {
        const unsigned short* qb = Qa + ((int64_t)(b * NM + mtile * 16 + l15)) * CH;
        const int off = half ? 16 : 0;
        for (int cc = 0; cc < 8; ++cc) {
            BFU u;
            __builtin_memcpy(&u.h[0], qb + cc * 32 + off, 16);
            __builtin_memcpy(&u.h[1], qb + cc * 32 + off + 8, 16);
            qf[cc] = u.v;
        }
    }

    float runmax[8], runsum[8];
    v8f acc[16];
    const v8f vzero = {};
    for (int r = 0; r < 8; ++r) { runmax[r] = -3.0e38f; runsum[r] = 0.f; }
    for (int t = 0; t < 16; ++t) acc[t] = vzero;

    const int nstart = split * NSEG;
    const int nend   = nstart + NSEG;

    for (int nb = nstart; nb < nend; nb += 32) {
        const unsigned short* kb0 = Kt + ((int64_t)(b * NN + nb + l15)) * CH + koff;
        const unsigned short* kb1 = kb0 + (int64_t)16 * CH;
        const unsigned short* vb  = Vt + ((int64_t)(b * CH + l15)) * NN + nb + koff;

        if (nb + 32 < nend) {
            __builtin_prefetch(kb0 + (int64_t)32 * CH, 0, 1);   // next K tile
            __builtin_prefetch(vb + 32, 0, 1);                  // next V tile
        }

        // ---- GEMM1: S(16x32) accumulated over 8 c-chunks, double-buffered loads
        v8f s0 = vzero, s1 = vzero;
        BFU ka, kb_, ka2, kb2;
        __builtin_memcpy(&ka.h[0],  kb0, 16);      __builtin_memcpy(&ka.h[1],  kb0 + 8, 16);
        __builtin_memcpy(&kb_.h[0], kb1, 16);      __builtin_memcpy(&kb_.h[1], kb1 + 8, 16);
        for (int cc = 0; cc < 8; ++cc) {
            if (cc < 7) {
                __builtin_memcpy(&ka2.h[0], kb0 + (cc + 1) * 32, 16);
                __builtin_memcpy(&ka2.h[1], kb0 + (cc + 1) * 32 + 8, 16);
                __builtin_memcpy(&kb2.h[0], kb1 + (cc + 1) * 32, 16);
                __builtin_memcpy(&kb2.h[1], kb1 + (cc + 1) * 32 + 8, 16);
            }
            s0 = __builtin_amdgcn_wmma_f32_16x16x32_bf16(false, qf[cc], false, ka.v,
                                                         (short)0, s0, false, false);
            s1 = __builtin_amdgcn_wmma_f32_16x16x32_bf16(false, qf[cc], false, kb_.v,
                                                         (short)0, s1, false, false);
            ka = ka2; kb_ = kb2;
        }

        // ---- online softmax: row r of this half -> reduce across 16-lane half
        float p0[8], p1[8], corr[8];
        for (int r = 0; r < 8; ++r) {
            float tm = fmaxf(s0[r], s1[r]);
            tm = fmaxf(tm, __shfl_xor(tm, 1, 16));
            tm = fmaxf(tm, __shfl_xor(tm, 2, 16));
            tm = fmaxf(tm, __shfl_xor(tm, 4, 16));
            tm = fmaxf(tm, __shfl_xor(tm, 8, 16));
            float nm = fmaxf(runmax[r], tm);
            float cr = __expf(runmax[r] - nm);
            p0[r] = __expf(s0[r] - nm);
            p1[r] = __expf(s1[r] - nm);
            float rs = p0[r] + p1[r];
            rs += __shfl_xor(rs, 1, 16);
            rs += __shfl_xor(rs, 2, 16);
            rs += __shfl_xor(rs, 4, 16);
            rs += __shfl_xor(rs, 8, 16);
            runsum[r] = runsum[r] * cr + rs;
            runmax[r] = nm;
            corr[r]   = cr;
        }
        for (int t = 0; t < 16; ++t)
            for (int r = 0; r < 8; ++r) acc[t][r] *= corr[r];

        // ---- re-layout P (C/D layout) -> A-fragment via LDS bounce (bf16)
        for (int r = 0; r < 8; ++r) {
            int row = r + half * 8;
            pt16[row * 32 + l15]      = f2bf(p0[r]);
            pt16[row * 32 + 16 + l15] = f2bf(p1[r]);
        }
        __syncthreads();
        v16bf pf;
        {
            const int off = half ? 16 : 0;
            BFU u;
            __builtin_memcpy(&u.h[0], &pt16[l15 * 32 + off], 16);
            __builtin_memcpy(&u.h[1], &pt16[l15 * 32 + off + 8], 16);
            pf = u.v;
        }
        __syncthreads();

        // ---- GEMM2: acc(16x256) += P(16x32) . V(32x256), one-ahead pipelined
        BFU va, va2;
        __builtin_memcpy(&va.h[0], vb, 16);
        __builtin_memcpy(&va.h[1], vb + 8, 16);
        for (int t = 0; t < 16; ++t) {
            if (t < 15) {
                const unsigned short* vp = vb + (int64_t)(t + 1) * 16 * NN;
                __builtin_memcpy(&va2.h[0], vp, 16);
                __builtin_memcpy(&va2.h[1], vp + 8, 16);
            }
            acc[t] = __builtin_amdgcn_wmma_f32_16x16x32_bf16(false, pf, false, va.v,
                                                             (short)0, acc[t], false, false);
            va = va2;
        }
    }

    // ---- write partial accumulator (coalesced over c) and per-row stats
    float* ab = accbuf + ((int64_t)(split * NTILES + tile) * 16) * 256;
    for (int t = 0; t < 16; ++t)
        for (int r = 0; r < 8; ++r)
            ab[(r + half * 8) * 256 + t * 16 + l15] = acc[t][r];

    if (l15 == 0) {
        float* sb = statbuf + (int64_t)(split * NTILES + tile) * 32;
        for (int r = 0; r < 8; ++r) {
            int row = r + half * 8;
            sb[row]      = runmax[r];
            sb[16 + row] = runsum[r];
        }
    }
}

// ---------------------------------------------------------------------------
// Kernel 5: merge SPLIT partials with flash rescaling, scatter into out.
// One 256-thread block per (b, mtile) tile. Reads coalesced over c, writes
// coalesced over consecutive hole positions via an LDS transpose.
// ---------------------------------------------------------------------------
__global__ __launch_bounds__(256)
void flash_combine(const float* __restrict__ accbuf,   // [SPLIT][512][16][256]
                   const float* __restrict__ statbuf,  // [SPLIT][512][32]
                   float* __restrict__ out)            // [8][768][4096]
{
    const int tile  = blockIdx.x;       // 0..511
    const int b     = tile >> 6;
    const int mtile = tile & 63;
    const int tid   = threadIdx.x;

    __shared__ float sF[SPLIT][16];     // per-row rescale factor / L
    __shared__ float stile[16 * 256];   // merged tile

    if (tid < 16) {
        float mx[SPLIT];
        float M = -3.0e38f;
        for (int s = 0; s < SPLIT; ++s) {
            mx[s] = statbuf[(int64_t)(s * NTILES + tile) * 32 + tid];
            M = fmaxf(M, mx[s]);
        }
        float L = 0.f;
        float f[SPLIT];
        for (int s = 0; s < SPLIT; ++s) {
            float e = __expf(mx[s] - M);
            f[s] = e;
            L += statbuf[(int64_t)(s * NTILES + tile) * 32 + 16 + tid] * e;
        }
        float inv = 1.0f / L;
        for (int s = 0; s < SPLIT; ++s) sF[s][tid] = f[s] * inv;
    }
    __syncthreads();

    {   // merge: lane dimension = consecutive c -> coalesced reads
        const int mloc = tid >> 4;      // 0..15
        const int cg   = tid & 15;      // 0..15
        for (int j = 0; j < 16; ++j) {
            int c = j * 16 + cg;
            float v = 0.f;
            for (int s = 0; s < SPLIT; ++s)
                v += accbuf[((int64_t)(s * NTILES + tile) * 16 + mloc) * 256 + c] * sF[s][mloc];
            stile[mloc * 256 + c] = v;
        }
    }
    __syncthreads();

    {   // scatter: lane dimension = consecutive m -> coalesced 64B stores
        const int mloc = tid & 15;
        const int cg   = tid >> 4;
        const int pos  = mpos(mtile * 16 + mloc);
        for (int j = 0; j < 16; ++j) {
            int c = cg + j * 16;
            out[((int64_t)(b * OUTC + CTOT + c)) * HW + pos] = stile[mloc * 256 + c];
        }
    }
}

// ---------------------------------------------------------------------------
// Launcher
// ---------------------------------------------------------------------------
extern "C" void kernel_launch(void* const* d_in, const int* in_sizes, int n_in,
                              void* d_out, int out_size, void* d_ws, size_t ws_size,
                              hipStream_t stream) {
    (void)in_sizes; (void)n_in; (void)out_size; (void)ws_size;
    const float* in  = (const float*)d_in[0];   // (8,512,64,64) f32
    float*       out = (float*)d_out;           // (8,768,64,64) f32

    // workspace carve-out (~62 MB)
    float*          invfm  = (float*)d_ws;                        // 8*1024 f32
    float*          invln  = invfm + BZ * NM;                     // 8*3072 f32
    unsigned short* Qa     = (unsigned short*)(invln + BZ * NN);  // 8*1024*256 bf16
    unsigned short* Kt     = Qa + (size_t)BZ * NM * CH;           // 8*3072*256 bf16
    unsigned short* Vt     = Kt + (size_t)BZ * NN * CH;           // 8*256*3072 bf16
    float*          accbuf = (float*)(Vt + (size_t)BZ * CH * NN); // SPLIT*512*16*256 f32
    float*          statbuf = accbuf + (size_t)SPLIT * NTILES * 16 * 256; // SPLIT*512*32 f32

    // 1) copy former/latter, zero shift plane
    {
        int64_t total4 = (int64_t)BZ * OUTC * (HW / 4);
        int grid = (int)((total4 + 255) / 256);
        prep_copy_zero<<<grid, 256, 0, stream>>>(in, out);
    }
    // 2) inverse norms
    fm_inv_norm<<<(BZ * NM + 255) / 256, 256, 0, stream>>>(in, invfm);
    ln_inv_norm<<<(BZ * NN + 255) / 256, 256, 0, stream>>>(in, invln);
    // 3) bf16 staging in WMMA layouts
    {
        int64_t tq = (int64_t)BZ * NM * CH;
        stage_q<<<(int)((tq + 255) / 256), 256, 0, stream>>>(in, invfm, Qa);
        int64_t tk = (int64_t)BZ * NN * CH;
        stage_k<<<(int)((tk + 255) / 256), 256, 0, stream>>>(in, invln, Kt);
        int64_t tv = (int64_t)BZ * CH * NN;
        stage_v<<<(int)((tv + 255) / 256), 256, 0, stream>>>(in, Vt);
    }
    // 4) fused WMMA flash attention partials (split-K for occupancy)
    {
        dim3 grid(NM / 16, BZ, SPLIT);   // 64 x 8 x 4 = 2048 waves
        flash_attn_part<<<grid, 32, 0, stream>>>(Qa, Kt, Vt, accbuf, statbuf);
    }
    // 5) merge partials + coalesced scatter into shift plane
    flash_combine<<<NTILES, 256, 0, stream>>>(accbuf, statbuf, out);
}
